// SampleNet_59631325938032
// MI455X (gfx1250) — matrix operand order repaired and checked by reading, not compiled, not run on hardware
//
#include <hip/hip_runtime.h>
#include <hip/hip_bf16.h>

// ---------------------------------------------------------------------------
// Problem constants (from reference):  x[B,N,C], C==K-dim of both GEMMs
// ---------------------------------------------------------------------------
constexpr int B_   = 32;
constexpr int N_   = 8192;
constexpr int C_   = 256;
constexpr int BN_  = B_ * N_;        // 262144 token rows
constexpr int K_   = 2048;           // n_sampled_token
constexpr int ROWS = 64;             // token rows per workgroup in fused kernel
constexpr int GRID1 = BN_ / ROWS;    // 4096 blocks

typedef __attribute__((ext_vector_type(16))) _Float16 v16h;
typedef __attribute__((ext_vector_type(8)))  float    v8f;
typedef __attribute__((ext_vector_type(4)))  float    f32x4;
typedef __attribute__((ext_vector_type(4)))  unsigned u32x4;

union H16 { v16h v; u32x4 q[2]; _Float16 h[16]; };

// CDNA5 async global->LDS copy (ASYNCcnt path). VDST = 32-bit LDS byte offset
// (low 32 bits of the flat shared-memory address), ADDR = 64-bit global addr.
__device__ __forceinline__ void async_g2l_b128(void* ldsDst, const void* gSrc) {
  const unsigned loff = (unsigned)(uintptr_t)ldsDst;
  asm volatile("global_load_async_to_lds_b128 %0, %1, off"
               :: "v"(loff), "v"(gSrc) : "memory");
}
__device__ __forceinline__ void wait_asynccnt0() {
  asm volatile("s_wait_asynccnt 0x0" ::: "memory");
}

// ---------------------------------------------------------------------------
// 1) Transpose + f16-convert weights: Wt[d][c] = (f16) W[c][d]
//    (lets the WMMA B-fragment be two contiguous 16B loads per lane)
// ---------------------------------------------------------------------------
__global__ __launch_bounds__(256)
void prep_weights(const float* __restrict__ W1, const float* __restrict__ W2,
                  _Float16* __restrict__ W1t, _Float16* __restrict__ W2t) {
  const int b = blockIdx.x;        // 0..511
  const int c = threadIdx.x;       // 0..255
  if (b < C_) {
    W1t[b * C_ + c] = (_Float16)W1[c * C_ + b];
  } else {
    const int d = b - C_;
    W2t[d * C_ + c] = (_Float16)W2[c * C_ + d];
  }
}

// ---------------------------------------------------------------------------
// 2) Fused per-64-row block:
//      async-stage x tile (64KB, contiguous) -> LDS
//      h1 = gelu(x@W1+b1)   [WMMA f16, f32 acc, h1 -> LDS f16]
//      h2 = h1@W2+b2        [WMMA from LDS; h2 overlays h1 region]
//      hn = LayerNorm(h2);  s = sigmoid((x+hn)@Wt + bt)   (x from LDS stage)
//      r  = s + gumbel(u) + 10*zn   (+ deterministic per-block sum/sumsq)
// Block: 256 threads = 8 wave32, wave grid 2(M)x4(N); tile 64 rows x 256 cols.
// ---------------------------------------------------------------------------
__global__ __launch_bounds__(256)
void fused_mlp_kernel(const float* __restrict__ x,
                      const float* __restrict__ u,
                      const float* __restrict__ zn,
                      const float* __restrict__ b1,
                      const float* __restrict__ b2,
                      const float* __restrict__ lng,
                      const float* __restrict__ lnb,
                      const float* __restrict__ wt,
                      const float* __restrict__ bt,
                      const _Float16* __restrict__ W1t,
                      const _Float16* __restrict__ W2t,
                      float* __restrict__ rout,
                      float* __restrict__ bSum,
                      float* __restrict__ bSumSq) {
  extern __shared__ char smem[];
  float*    xs = (float*)smem;                       // 64x256 f32 stage (64 KB)
  _Float16* h1 = (_Float16*)(smem + 65536);          // 64x256 f16 (32 KB) ...
  float*    h2 = (float*)(smem + 65536);             // ... overlaid by 64 KB f32
  float*  redA = (float*)(smem + 131072);
  float*  redB = redA + 256;
  float*  muA  = redB + 256;
  float*  rsA  = muA + ROWS;
  float*  rbuf = rsA + ROWS;

  const int tid  = threadIdx.x;
  const int lane = tid & 31;
  const int wave = tid >> 5;
  const int mw   = wave >> 2;     // 0..1  (M wave)
  const int nw   = wave & 3;      // 0..3  (N wave)
  const int llo  = lane & 15;
  const int lhi  = lane >> 4;

  const size_t rowBlock = (size_t)blockIdx.x * ROWS;

  // ---- async stage: 64 contiguous rows of x (64 KB) into LDS --------------
  {
    const float* gx = x + rowBlock * C_;             // contiguous 64 KB
#pragma unroll
    for (int i = 0; i < (ROWS * C_) / (4 * 256); ++i) {   // 16 b128 per thread
      const int e4 = i * 256 + tid;                  // float4 index
      async_g2l_b128(xs + e4 * 4, gx + e4 * 4);
    }
    wait_asynccnt0();
  }
  __syncthreads();

  const v8f vzero = {};
  v8f acc[2][4];
#pragma unroll
  for (int i = 0; i < 2; ++i)
#pragma unroll
    for (int j = 0; j < 4; ++j) acc[i][j] = vzero;

  // ---------------- GEMM1: xs(LDS f32->f16) @ W1t, gelu -> h1 (LDS f16) ----
#pragma unroll 1
  for (int kb = 0; kb < 8; ++kb) {
    const int k0 = kb * 32 + lhi * 8;   // A halves 0..7 = K k0..k0+7, 8..15 = k0+16..+23
    H16 a[2];
#pragma unroll
    for (int mi = 0; mi < 2; ++mi) {
      const int m = mw * 32 + mi * 16 + llo;
      const float* px = xs + m * C_ + k0;
      f32x4 f0 = *(const f32x4*)(px);
      f32x4 f1 = *(const f32x4*)(px + 4);
      f32x4 f2 = *(const f32x4*)(px + 16);
      f32x4 f3 = *(const f32x4*)(px + 20);
#pragma unroll
      for (int e = 0; e < 4; ++e) {
        a[mi].h[e]      = (_Float16)f0[e];
        a[mi].h[4 + e]  = (_Float16)f1[e];
        a[mi].h[8 + e]  = (_Float16)f2[e];
        a[mi].h[12 + e] = (_Float16)f3[e];
      }
    }
    const int kc = kb * 32 + lhi * 8;   // B halves 0..7 = K kc..kc+7, 8..15 = kc+16..+23
    H16 bf[4];
#pragma unroll
    for (int ni = 0; ni < 4; ++ni) {
      const int col = nw * 64 + ni * 16 + llo;
      const u32x4* q = (const u32x4*)(W1t + (size_t)col * C_ + kc);
      bf[ni].q[0] = q[0];
      bf[ni].q[1] = q[2];
    }
#pragma unroll
    for (int mi = 0; mi < 2; ++mi)
#pragma unroll
      for (int ni = 0; ni < 4; ++ni)
        acc[mi][ni] = __builtin_amdgcn_wmma_f32_16x16x32_f16(
            false, a[mi].v, false, bf[ni].v, (short)0, acc[mi][ni], false, false);
  }
  // epilogue 1: +b1, exact gelu, store f16 to LDS; reset accumulators
#pragma unroll
  for (int mi = 0; mi < 2; ++mi)
#pragma unroll
    for (int ni = 0; ni < 4; ++ni) {
      const int n = nw * 64 + ni * 16 + llo;
      const float bias = b1[n];
#pragma unroll
      for (int j = 0; j < 8; ++j) {
        const int m = mw * 32 + mi * 16 + lhi * 8 + j;   // D layout: M = j + 8*(lane/16)
        const float v = acc[mi][ni][j] + bias;
        const float g = 0.5f * v * (1.0f + erff(v * 0.70710678118654752f));
        h1[m * C_ + n] = (_Float16)g;
      }
      acc[mi][ni] = vzero;
    }
  __syncthreads();

  // ---------------- GEMM2: h1(LDS f16) @ W2t ------------------------------
#pragma unroll 1
  for (int kb = 0; kb < 8; ++kb) {
    const int k0 = kb * 32 + lhi * 8;
    H16 a[2];
#pragma unroll
    for (int mi = 0; mi < 2; ++mi) {
      const int m = mw * 32 + mi * 16 + llo;
      a[mi].q[0] = *(const u32x4*)(h1 + m * C_ + k0);        // ds_load_b128
      a[mi].q[1] = *(const u32x4*)(h1 + m * C_ + k0 + 16);   // ds_load_b128
    }
    const int kc = kb * 32 + lhi * 8;
    H16 bf[4];
#pragma unroll
    for (int ni = 0; ni < 4; ++ni) {
      const int col = nw * 64 + ni * 16 + llo;
      const u32x4* q = (const u32x4*)(W2t + (size_t)col * C_ + kc);
      bf[ni].q[0] = q[0];
      bf[ni].q[1] = q[2];
    }
#pragma unroll
    for (int mi = 0; mi < 2; ++mi)
#pragma unroll
      for (int ni = 0; ni < 4; ++ni)
        acc[mi][ni] = __builtin_amdgcn_wmma_f32_16x16x32_f16(
            false, a[mi].v, false, bf[ni].v, (short)0, acc[mi][ni], false, false);
  }
  __syncthreads();   // all h1 reads done: h2 may now overlay the h1 region

  // epilogue 2: +b2 -> h2 (f32 LDS, overlays h1)
#pragma unroll
  for (int mi = 0; mi < 2; ++mi)
#pragma unroll
    for (int ni = 0; ni < 4; ++ni) {
      const int n = nw * 64 + ni * 16 + llo;
      const float bias = b2[n];
#pragma unroll
      for (int j = 0; j < 8; ++j) {
        const int m = mw * 32 + mi * 16 + lhi * 8 + j;
        h2[m * C_ + n] = acc[mi][ni][j] + bias;
      }
    }
  __syncthreads();

  // ---------------- LayerNorm + score + noise ------------------------------
  const int row  = tid >> 2;         // 0..63
  const int part = tid & 3;          // 0..3 (64 cols each)
  const float* hr = h2 + row * C_ + part * 64;
  float s1 = 0.0f, s2 = 0.0f;
#pragma unroll 4
  for (int c = 0; c < 64; ++c) { const float v = hr[c]; s1 += v; s2 += v * v; }
  redA[tid] = s1; redB[tid] = s2;
  __syncthreads();
  if (part == 0) {
    const float S  = redA[tid] + redA[tid + 1] + redA[tid + 2] + redA[tid + 3];
    const float SS = redB[tid] + redB[tid + 1] + redB[tid + 2] + redB[tid + 3];
    const float mu  = S * (1.0f / C_);
    const float var = SS * (1.0f / C_) - mu * mu;   // biased, as reference
    muA[row] = mu;
    rsA[row] = rsqrtf(var + 1e-5f);
  }
  __syncthreads();
  const float mu = muA[row];
  const float rs = rsA[row];
  const size_t gRow = rowBlock + row;
  const float* xr = xs + row * C_ + part * 64;      // x from LDS stage
  float tacc = 0.0f;
#pragma unroll 4
  for (int c = 0; c < 64; ++c) {
    const int cc = part * 64 + c;
    const float hn = (hr[c] - mu) * rs * lng[cc] + lnb[cc];
    tacc += (xr[c] + hn) * wt[cc];
  }
  redA[tid] = tacc;
  __syncthreads();
  if (part == 0) {
    const float t = redA[tid] + redA[tid + 1] + redA[tid + 2] + redA[tid + 3] + bt[0];
    const float s = 1.0f / (1.0f + expf(-t));
    const float uu = u[gRow];
    const float gum = -logf(-logf(uu + 1e-20f) + 1e-20f);
    const float rv = s + gum + 10.0f * zn[gRow];
    rout[gRow] = rv;
    rbuf[row]  = rv;
  }
  __syncthreads();
  if (tid == 0) {                         // deterministic per-block partials
    float S = 0.0f, SS = 0.0f;
    for (int i = 0; i < ROWS; ++i) { const float v = rbuf[i]; S += v; SS += v * v; }
    bSum[blockIdx.x]   = S;
    bSumSq[blockIdx.x] = SS;
  }
}

// ---------------------------------------------------------------------------
// 3) Global mean / invstd (ddof=1) from 4096 deterministic partials
// ---------------------------------------------------------------------------
__global__ __launch_bounds__(1024)
void reduce_stats(const float* __restrict__ bSum, const float* __restrict__ bSumSq,
                  float* __restrict__ stats) {
  __shared__ float sA[1024], sB[1024];
  const int tid = threadIdx.x;
  float a = 0.0f, b = 0.0f;
  for (int i = tid; i < GRID1; i += 1024) { a += bSum[i]; b += bSumSq[i]; }
  sA[tid] = a; sB[tid] = b;
  __syncthreads();
  for (int off = 512; off; off >>= 1) {
    if (tid < off) { sA[tid] += sA[tid + off]; sB[tid] += sB[tid + off]; }
    __syncthreads();
  }
  if (tid == 0) {
    const double S = sA[0], SS = sB[0], M = (double)BN_;
    const double mean = S / M;
    const double var  = (SS - S * S / M) / (M - 1.0);   // ddof=1
    stats[0] = (float)mean;
    stats[1] = (float)(1.0 / sqrt(var));
  }
}

// ---------------------------------------------------------------------------
// 4) Per-batch softmax stats of z = 2*(r-mean)*invstd  (temperature 0.5)
// ---------------------------------------------------------------------------
__global__ __launch_bounds__(256)
void softmax_stats(const float* __restrict__ r, const float* __restrict__ stats,
                   float* __restrict__ rowMax, float* __restrict__ rowSum) {
  __shared__ float red[256];
  const int b = blockIdx.x, tid = threadIdx.x;
  const float mean = stats[0], istd = stats[1];
  const float* rb = r + (size_t)b * N_;
  float m = -INFINITY;
  for (int i = tid; i < N_; i += 256)
    m = fmaxf(m, 2.0f * (rb[i] - mean) * istd);
  red[tid] = m;
  __syncthreads();
  for (int off = 128; off; off >>= 1) {
    if (tid < off) red[tid] = fmaxf(red[tid], red[tid + off]);
    __syncthreads();
  }
  const float M = red[0];
  __syncthreads();
  float s = 0.0f;
  for (int i = tid; i < N_; i += 256)
    s += expf(2.0f * (rb[i] - mean) * istd - M);
  red[tid] = s;
  __syncthreads();
  for (int off = 128; off; off >>= 1) {
    if (tid < off) red[tid] += red[tid + off];
    __syncthreads();
  }
  if (tid == 0) { rowMax[b] = M; rowSum[b] = red[0]; }
}

// ---------------------------------------------------------------------------
// 5) Per-batch top-K radix select (MSB-first, 4 passes) + stable ordered
//    compaction -> indices already sorted ascending. Integer LDS atomics only.
// ---------------------------------------------------------------------------
__global__ __launch_bounds__(1024)
void topk_select(const float* __restrict__ r, int* __restrict__ idxOut) {
  __shared__ unsigned keys[N_];     // 32 KB
  __shared__ unsigned hist[256];
  __shared__ unsigned scan[1024];
  __shared__ unsigned info[2];
  const int b = blockIdx.x, tid = threadIdx.x;
  const float* rb = r + (size_t)b * N_;
  for (int i = tid; i < N_; i += 1024) {
    const unsigned bits = __float_as_uint(rb[i]);
    keys[i] = (bits & 0x80000000u) ? ~bits : (bits | 0x80000000u);  // order-preserving
  }
  __syncthreads();

  unsigned prefix = 0u, remaining = (unsigned)K_;
  const int base = tid * 8;                       // contiguous -> stable ordering
  for (int pass = 0; pass < 4; ++pass) {
    const int sh = 24 - 8 * pass;
    if (tid < 256) hist[tid] = 0u;
    __syncthreads();
#pragma unroll
    for (int e = 0; e < 8; ++e) {
      const unsigned k = keys[base + e];
      const bool ok = (pass == 0) || ((k >> (sh + 8)) == prefix);
      if (ok) atomicAdd(&hist[(k >> sh) & 255u], 1u);
    }
    __syncthreads();
    if (tid == 0) {
      unsigned cum = 0u;
      for (int bb = 255; bb >= 0; --bb) {
        const unsigned c = hist[bb];
        if (cum + c >= remaining) { info[0] = (unsigned)bb; info[1] = remaining - cum; break; }
        cum += c;
      }
    }
    __syncthreads();
    prefix = (prefix << 8) | info[0];
    remaining = info[1];
    __syncthreads();
  }
  const unsigned T = prefix;        // K-th largest key value
  const unsigned R = remaining;     // ties at T to keep (first R in index order)

  unsigned kk[8];
  unsigned myE = 0u;
#pragma unroll
  for (int e = 0; e < 8; ++e) { kk[e] = keys[base + e]; myE += (kk[e] == T); }

  // inclusive scan of equal-counts (index order preserved by contiguous chunks)
  unsigned val = myE;
  scan[tid] = val;
  __syncthreads();
  for (int off = 1; off < 1024; off <<= 1) {
    const unsigned t = (tid >= off) ? scan[tid - off] : 0u;
    __syncthreads();
    val += t; scan[tid] = val;
    __syncthreads();
  }
  unsigned eqr = val - myE;         // equals-before-me (exclusive)

  unsigned flags = 0u, mySel = 0u;
#pragma unroll
  for (int e = 0; e < 8; ++e) {
    bool s = false;
    if (kk[e] > T) s = true;
    else if (kk[e] == T) { s = (eqr < R); ++eqr; }
    flags |= ((unsigned)s) << e;
    mySel += (unsigned)s;
  }
  __syncthreads();
  // inclusive scan of selected-counts -> stable write positions (sorted idx)
  val = mySel;
  scan[tid] = val;
  __syncthreads();
  for (int off = 1; off < 1024; off <<= 1) {
    const unsigned t = (tid >= off) ? scan[tid - off] : 0u;
    __syncthreads();
    val += t; scan[tid] = val;
    __syncthreads();
  }
  unsigned pos = val - mySel;
  int* ob = idxOut + (size_t)b * K_;
#pragma unroll
  for (int e = 0; e < 8; ++e)
    if ((flags >> e) & 1u) ob[pos++] = base + e;
}

// ---------------------------------------------------------------------------
// 6) Gather: q_sel[b,k,:] = x[b, idx, :] * ((1-p)+p),  p from softmax stats
// ---------------------------------------------------------------------------
__global__ __launch_bounds__(256)
void gather_scale(const float* __restrict__ x, const float* __restrict__ r,
                  const float* __restrict__ stats, const float* __restrict__ rowMax,
                  const float* __restrict__ rowSum, const int* __restrict__ idx,
                  float* __restrict__ out) {
  const int b  = blockIdx.y;
  const int kk = blockIdx.x * 4 + (threadIdx.x >> 6);   // 4 tokens per block
  const int l  = threadIdx.x & 63;                      // 64 lanes * float4 = 256
  const int i  = idx[(size_t)b * K_ + kk];
  const float mean = stats[0], istd = stats[1];
  const float z = 2.0f * (r[(size_t)b * N_ + i] - mean) * istd;
  const float p = expf(z - rowMax[b]) / rowSum[b];
  const float ret = (1.0f - p) + p;                     // reference op order
  const f32x4* xs = (const f32x4*)(x + ((size_t)b * N_ + i) * C_);
  f32x4* o = (f32x4*)(out + ((size_t)b * K_ + kk) * C_);
  f32x4 v = xs[l];
  v *= ret;
  o[l] = v;
}

// ---------------------------------------------------------------------------
// Launch
// ---------------------------------------------------------------------------
extern "C" void kernel_launch(void* const* d_in, const int* in_sizes, int n_in,
                              void* d_out, int out_size, void* d_ws, size_t ws_size,
                              hipStream_t stream) {
  const float* x   = (const float*)d_in[0];
  const float* u   = (const float*)d_in[1];
  const float* zn  = (const float*)d_in[2];
  const float* W1  = (const float*)d_in[3];
  const float* b1  = (const float*)d_in[4];
  const float* W2  = (const float*)d_in[5];
  const float* b2  = (const float*)d_in[6];
  const float* lng = (const float*)d_in[7];
  const float* lnb = (const float*)d_in[8];
  const float* wt  = (const float*)d_in[9];
  const float* bt  = (const float*)d_in[10];

  char* ws = (char*)d_ws;
  _Float16* W1t = (_Float16*)(ws);                        // 128 KB
  _Float16* W2t = (_Float16*)(ws + 131072);               // 128 KB
  float* r      = (float*)(ws + 262144);                  // 1 MB
  float* bSum   = (float*)(ws + 262144 + (size_t)BN_ * 4);
  float* bSumSq = bSum + GRID1;
  float* stats  = bSumSq + GRID1;
  float* rowMax = stats + 16;
  float* rowSum = rowMax + 64;

  float* q_out = (float*)d_out;
  int* idxOut  = (int*)d_out + (size_t)B_ * K_ * C_;      // idx after q_sel payload

  // LDS: 64KB x-stage + 64KB h1/h2 union + reduction scratch
  const size_t shmem = 65536 + 65536 +
                       (256 + 256 + ROWS + ROWS + ROWS) * sizeof(float); // 133888 B
  (void)hipFuncSetAttribute((const void*)fused_mlp_kernel,
                            hipFuncAttributeMaxDynamicSharedMemorySize, (int)shmem);

  prep_weights<<<dim3(512), dim3(256), 0, stream>>>(W1, W2, W1t, W2t);
  fused_mlp_kernel<<<dim3(GRID1), dim3(256), shmem, stream>>>(
      x, u, zn, b1, b2, lng, lnb, wt, bt, W1t, W2t, r, bSum, bSumSq);
  reduce_stats<<<dim3(1), dim3(1024), 0, stream>>>(bSum, bSumSq, stats);
  softmax_stats<<<dim3(B_), dim3(256), 0, stream>>>(r, stats, rowMax, rowSum);
  topk_select<<<dim3(B_), dim3(1024), 0, stream>>>(r, idxOut);
  gather_scale<<<dim3(K_ / 4, B_), dim3(256), 0, stream>>>(
      x, r, stats, rowMax, rowSum, idxOut, q_out);
}
